// MambaMoEGRUAttentionBlock_69501160784178
// MI455X (gfx1250) — matrix operand order repaired and checked by reading, not compiled, not run on hardware
//
#include <hip/hip_runtime.h>
#include <hip/hip_bf16.h>
#include <math.h>

// Problem constants (from reference): B=8, L=1024, D=128, DI=256, N=16, K=4,
// E=4, NH=4, dh=32, DTR=8, hidden=512, DEPTH=2.
#define BB   8
#define LLEN 1024
#define DD   128
#define DI_  256
#define NST  16
#define NEXP 4
#define DHH  32
#define TOK  (BB * LLEN)

typedef __attribute__((ext_vector_type(16))) _Float16 v16h;
typedef __attribute__((ext_vector_type(8)))  _Float16 v8h;
typedef __attribute__((ext_vector_type(8)))  float    v8f;

__device__ __forceinline__ float sigmoidf_(float x) { return 1.0f / (1.0f + __expf(-x)); }
__device__ __forceinline__ float siluf_(float x)    { return x * sigmoidf_(x); }
__device__ __forceinline__ float geluf_(float x)    { return 0.5f * x * (1.0f + erff(x * 0.70710678118654752f)); }
__device__ __forceinline__ float softplusf_(float x){ return (x > 20.0f) ? x : log1pf(__expf(x)); }

// ---------------------------------------------------------------------------
// f32 -> f16 conversion kernels
// ---------------------------------------------------------------------------
__global__ void cvt_f16_kernel(const float* __restrict__ src, _Float16* __restrict__ dst, int n) {
  int t = blockIdx.x * blockDim.x + threadIdx.x;
  if (t < n) dst[t] = (_Float16)src[t];
}

// Produce weight in (N,K) row-major f16. srcTrans=0: src is (K,N); 1: src is (N,K).
__global__ void cvt_w_kernel(const float* __restrict__ src, _Float16* __restrict__ dst,
                             int Nn, int Kk, int srcTrans) {
  int t = blockIdx.x * blockDim.x + threadIdx.x;
  if (t >= Nn * Kk) return;
  int n = t / Kk, k = t % Kk;
  dst[t] = (_Float16)(srcTrans ? src[(size_t)n * Kk + k] : src[(size_t)k * Nn + n]);
}

// ---------------------------------------------------------------------------
// WMMA GEMM: out[M,N] = epilogue( alpha * A[M,K](f16) * W^T + bias )
// W is (N,K)-logical f16 with row stride ldw (so fragments can be sliced out
// of packed buffers like qkv). Block = 128 threads = 4 waves; each wave does a
// 16x32 tile: one A fragment feeds two independent wmma per K-step.
// act: 0=none 1=gelu 2=silu ; gate: per-row scale gate[m*gstride]
// Output: Ch!=null -> f16 store; else f32 store/accumulate per `accum`.
// Requires M%64==0, N%32==0, K%32==0 (true for every call site).
// ---------------------------------------------------------------------------
__global__ __launch_bounds__(128)
void gemm_wmma_kernel(const _Float16* __restrict__ A, int lda,
                      const _Float16* __restrict__ W, int ldw,
                      const float* __restrict__ bias,
                      float* __restrict__ C, _Float16* __restrict__ Ch, int ldc,
                      int M, int N, int K,
                      int act, const float* __restrict__ gate, int gstride,
                      int accum, float alpha) {
  const int wave = threadIdx.x >> 5;
  const int lane = threadIdx.x & 31;
  const int tile_m = (blockIdx.x * 4 + wave) * 16;
  const int tile_n = blockIdx.y * 32;
  if (tile_m >= M) return;                 // wave-uniform

  const int half = lane >> 4;              // lane half selects K sub-range
  const int mr   = lane & 15;

  // A fragment rows: element i -> k = k0 + half*16 + i
  const _Float16* arow  = A + (size_t)(tile_m + mr) * lda + half * 16;
  // B fragment rows: element i -> k = k0 + half*16 + i
  const _Float16* wrow0 = W + (size_t)(tile_n + mr) * ldw + half * 16;
  const _Float16* wrow1 = W + (size_t)(tile_n + 16 + mr) * ldw + half * 16;

  v8f acc0 = {}, acc1 = {};
  for (int k0 = 0; k0 < K; k0 += 32) {
    union { v16h v; v8h h[2]; } ua;
    ua.h[0] = *(const v8h*)(arow + k0);            // 16B aligned
    ua.h[1] = *(const v8h*)(arow + k0 + 8);
    v16h b0 = *(const v16h*)(wrow0 + k0);          // 32B aligned
    v16h b1 = *(const v16h*)(wrow1 + k0);
    if (k0 + 32 < K)                               // global_prefetch_b8 next A chunk
      __builtin_prefetch((const void*)(arow + k0 + 32), 0, 3);
    acc0 = __builtin_amdgcn_wmma_f32_16x16x32_f16(false, ua.v, false, b0,
                                                  (short)0, acc0, false, false);
    acc1 = __builtin_amdgcn_wmma_f32_16x16x32_f16(false, ua.v, false, b1,
                                                  (short)0, acc1, false, false);
  }

  const int n0 = tile_n + mr, n1 = n0 + 16;
  const float bv0 = bias ? bias[n0] : 0.0f;
  const float bv1 = bias ? bias[n1] : 0.0f;
  #pragma unroll
  for (int r = 0; r < 8; ++r) {
    const int m = tile_m + r + half * 8;           // C/D layout: VGPR r -> M=r (+8 upper half)
    float v0 = acc0[r] * alpha + bv0;
    float v1 = acc1[r] * alpha + bv1;
    if (act == 1)      { v0 = geluf_(v0); v1 = geluf_(v1); }
    else if (act == 2) { v0 = siluf_(v0); v1 = siluf_(v1); }
    if (gate) { float gv = gate[(size_t)m * gstride]; v0 *= gv; v1 *= gv; }
    if (Ch) {
      Ch[(size_t)m * ldc + n0] = (_Float16)v0;
      Ch[(size_t)m * ldc + n1] = (_Float16)v1;
    } else {
      float* o0 = C + (size_t)m * ldc + n0;
      float* o1 = C + (size_t)m * ldc + n1;
      if (accum) { *o0 += v0; *o1 += v1; } else { *o0 = v0; *o1 = v1; }
    }
  }
}

// ---------------------------------------------------------------------------
// Mamba pieces
// ---------------------------------------------------------------------------
__global__ void conv_silu_kernel(const float* __restrict__ xr,   // (TOK,512), xc in [:,:256]
                                 const float* __restrict__ w,    // (256,1,4)
                                 const float* __restrict__ cb,   // (256,)
                                 float* __restrict__ u) {        // (TOK,256)
  int t = blockIdx.x * blockDim.x + threadIdx.x;
  if (t >= TOK * DI_) return;
  int d = t & 255, tok = t >> 8, l = tok & (LLEN - 1);
  float s = cb[d];
  #pragma unroll
  for (int k = 0; k < 4; ++k) {
    int ll = l - 3 + k;                       // causal left pad (K-1,0)
    if (ll >= 0) s += w[d * 4 + k] * xr[(size_t)(tok - 3 + k) * 512 + d];
  }
  u[t] = siluf_(s);
}

__global__ void xd_kernel(const float* __restrict__ u,     // (TOK,256)
                          const float* __restrict__ Wx,    // (256,40)
                          float* __restrict__ xd) {        // (TOK,40)
  int t = blockIdx.x * blockDim.x + threadIdx.x;
  if (t >= TOK * 40) return;
  int e = t % 40; size_t tok = t / 40;
  const float* ur = u + tok * 256;
  float s = 0.0f;
  for (int d = 0; d < 256; ++d) s += ur[d] * Wx[(size_t)d * 40 + e];
  xd[tok * 40 + e] = s;
}

__global__ void delta_kernel(const float* __restrict__ xd,   // (TOK,40), uses [:8]
                             const float* __restrict__ Wdt,  // (8,256)
                             const float* __restrict__ bdt,  // (256,)
                             float* __restrict__ delta) {    // (TOK,256)
  int t = blockIdx.x * blockDim.x + threadIdx.x;
  if (t >= TOK * DI_) return;
  int d = t & 255; size_t tok = t >> 8;
  const float* xr = xd + tok * 40;
  float s = bdt[d];
  #pragma unroll
  for (int k = 0; k < 8; ++k) s += xr[k] * Wdt[k * 256 + d];
  delta[t] = softplusf_(s);
}

// Selective scan: one thread per (b,d); 16-state recurrence in registers.
__global__ void scan_kernel(const float* __restrict__ delta, // (TOK,256)
                            const float* __restrict__ u,     // (TOK,256)
                            const float* __restrict__ xd,    // (TOK,40): B=[8:24], C=[24:40]
                            const float* __restrict__ A_log, // (256,16)
                            float* __restrict__ y) {         // (TOK,256)
  int tid = blockIdx.x * blockDim.x + threadIdx.x;           // 0..2047
  if (tid >= BB * DI_) return;
  int b = tid >> 8, d = tid & 255;
  float Av[NST], s[NST];
  #pragma unroll
  for (int n = 0; n < NST; ++n) { Av[n] = -__expf(A_log[d * NST + n]); s[n] = 0.0f; }
  for (int l = 0; l < LLEN; ++l) {
    size_t tok = (size_t)b * LLEN + l;
    float dt = delta[tok * 256 + d];
    float uu = u[tok * 256 + d];
    const float* xdr = xd + tok * 40;
    float yy = 0.0f;
    #pragma unroll
    for (int n = 0; n < NST; ++n) {
      float dA = __expf(dt * Av[n]);
      s[n] = dA * s[n] + dt * xdr[8 + n] * uu;
      yy += s[n] * xdr[24 + n];
    }
    y[tok * 256 + d] = yy;
  }
}

__global__ void gate_y_kernel(const float* __restrict__ yin,
                              const float* __restrict__ u,
                              const float* __restrict__ xr,   // (TOK,512), res in [:,256:]
                              const float* __restrict__ Dp,
                              float* __restrict__ yout) {
  int t = blockIdx.x * blockDim.x + threadIdx.x;
  if (t >= TOK * DI_) return;
  int d = t & 255; size_t tok = t >> 8;
  float res = xr[tok * 512 + 256 + d];
  yout[t] = (yin[t] + u[t] * Dp[d]) * siluf_(res);
}

__global__ void reverse_copy_kernel(const float* __restrict__ in, float* __restrict__ out) {
  int t = blockIdx.x * blockDim.x + threadIdx.x;
  if (t >= TOK * DD) return;
  int c = t & 127, tok = t >> 7, l = tok & (LLEN - 1), b = tok >> 10;
  out[t] = in[((size_t)b * LLEN + (LLEN - 1 - l)) * DD + c];
}

__global__ void add_reverse_kernel(float* __restrict__ acc, const float* __restrict__ in) {
  int t = blockIdx.x * blockDim.x + threadIdx.x;
  if (t >= TOK * DD) return;
  int c = t & 127, tok = t >> 7, l = tok & (LLEN - 1), b = tok >> 10;
  acc[t] += in[((size_t)b * LLEN + (LLEN - 1 - l)) * DD + c];
}

// ---------------------------------------------------------------------------
// LayerNorm over D=128: one wave per token (wave32 -> 4 elems/lane)
// ---------------------------------------------------------------------------
__global__ __launch_bounds__(256)
void ln_kernel(const float* __restrict__ in1, const float* __restrict__ in2,
               const float* __restrict__ g, const float* __restrict__ bta,
               float* __restrict__ out) {
  int wave = threadIdx.x >> 5, lane = threadIdx.x & 31;
  int tok = blockIdx.x * 8 + wave;
  const float* p1 = in1 + (size_t)tok * DD;
  const float* p2 = in2 + (size_t)tok * DD;
  float v[4], s = 0.0f;
  #pragma unroll
  for (int j = 0; j < 4; ++j) { float t = p1[lane * 4 + j] + p2[lane * 4 + j]; v[j] = t; s += t; }
  #pragma unroll
  for (int o = 16; o; o >>= 1) s += __shfl_xor(s, o, 32);
  float mean = s * (1.0f / 128.0f);
  float var = 0.0f;
  #pragma unroll
  for (int j = 0; j < 4; ++j) { float t = v[j] - mean; var += t * t; }
  #pragma unroll
  for (int o = 16; o; o >>= 1) var += __shfl_xor(var, o, 32);
  float inv = rsqrtf(var * (1.0f / 128.0f) + 1e-5f);
  float* po = out + (size_t)tok * DD;
  #pragma unroll
  for (int j = 0; j < 4; ++j)
    po[lane * 4 + j] = (v[j] - mean) * inv * g[lane * 4 + j] + bta[lane * 4 + j];
}

// ---------------------------------------------------------------------------
// MoE routing (top-1 with batch-axis capacity normalization, as in reference)
// ---------------------------------------------------------------------------
__global__ void moe_gate1_kernel(const float* __restrict__ x, const float* __restrict__ Wg,
                                 const float* __restrict__ bg, float* __restrict__ gates) {
  int tok = blockIdx.x * blockDim.x + threadIdx.x;
  if (tok >= TOK) return;
  const float* xr = x + (size_t)tok * DD;
  float lg[NEXP], mx = -1e30f;
  for (int e = 0; e < NEXP; ++e) {
    float s = bg[e];
    for (int d = 0; d < DD; ++d) s += xr[d] * Wg[d * NEXP + e];
    lg[e] = s; mx = fmaxf(mx, s);
  }
  float sum = 0.0f;
  for (int e = 0; e < NEXP; ++e) { lg[e] = __expf(lg[e] - mx); sum += lg[e]; }
  int am = 0; float best = lg[0];
  for (int e = 1; e < NEXP; ++e) if (lg[e] > best) { best = lg[e]; am = e; }
  for (int e = 0; e < NEXP; ++e) gates[(size_t)tok * NEXP + e] = (e == am) ? lg[e] / sum : 0.0f;
}

__global__ void moe_gate2_kernel(float* __restrict__ gates) {   // normalize over batch per (l,e)
  int t = blockIdx.x * blockDim.x + threadIdx.x;
  if (t >= LLEN * NEXP) return;
  int l = t >> 2, e = t & 3;
  float denom = 1e-6f;
  for (int b = 0; b < BB; ++b) denom += gates[((size_t)b * LLEN + l) * NEXP + e];
  float sc = (float)BB / denom;                                  // capacity = B = 8
  for (int b = 0; b < BB; ++b) gates[((size_t)b * LLEN + l) * NEXP + e] *= sc;
}

// ---------------------------------------------------------------------------
// GRU: single workgroup, 1024 threads; sequential over L with h/gh in LDS
// ---------------------------------------------------------------------------
__global__ __launch_bounds__(1024)
void gru_kernel(const float* __restrict__ gx,   // (TOK,384) = x@Wih^T + bih
                const float* __restrict__ Whh,  // (384,128)
                const float* __restrict__ bhh,  // (384,)
                float* __restrict__ Hout) {     // (TOK,128)
  __shared__ float h[BB * DD];       // 4 KB
  __shared__ float gh[BB * 384];     // 12 KB
  int tid = threadIdx.x;
  h[tid] = 0.0f;
  __syncthreads();
  for (int l = 0; l < LLEN; ++l) {
    for (int idx = tid; idx < BB * 384; idx += 1024) {
      int b = idx / 384, j = idx % 384;
      const float* hrow = &h[b * DD];
      const float* wrow = &Whh[(size_t)j * DD];
      float s = bhh[j];
      #pragma unroll 4
      for (int d = 0; d < DD; ++d) s += hrow[d] * wrow[d];
      gh[idx] = s;
    }
    __syncthreads();
    {
      int b = tid >> 7, d = tid & 127;
      const float* g = &gx[((size_t)b * LLEN + l) * 384];
      float r  = sigmoidf_(g[d]       + gh[b * 384 + d]);
      float z  = sigmoidf_(g[128 + d] + gh[b * 384 + 128 + d]);
      float nn = tanhf   (g[256 + d] + r * gh[b * 384 + 256 + d]);
      float hv = (1.0f - z) * nn + z * h[tid];
      h[tid] = hv;
      Hout[((size_t)b * LLEN + l) * DD + d] = hv;
    }
    __syncthreads();
  }
}

// ---------------------------------------------------------------------------
// Attention helpers: row softmax (f32 scores -> f16 probs) and V transpose
// ---------------------------------------------------------------------------
__global__ __launch_bounds__(256)
void row_softmax_f16_kernel(const float* __restrict__ S, _Float16* __restrict__ P) {
  int wave = threadIdx.x >> 5, lane = threadIdx.x & 31;
  int row = blockIdx.x * 8 + wave;
  const float* sr = S + (size_t)row * LLEN;
  float mx = -1e30f;
  for (int k = lane; k < LLEN; k += 32) mx = fmaxf(mx, sr[k]);
  #pragma unroll
  for (int o = 16; o; o >>= 1) mx = fmaxf(mx, __shfl_xor(mx, o, 32));
  float sum = 0.0f;
  for (int k = lane; k < LLEN; k += 32) sum += __expf(sr[k] - mx);
  #pragma unroll
  for (int o = 16; o; o >>= 1) sum += __shfl_xor(sum, o, 32);
  float inv = 1.0f / sum;
  _Float16* pr = P + (size_t)row * LLEN;
  for (int k = lane; k < LLEN; k += 32) pr[k] = (_Float16)(__expf(sr[k] - mx) * inv);
}

// vt[bh][d][k] = qkv_f16[(b*L+k)*384 + 256 + h*32 + d]
__global__ void vT_kernel(const _Float16* __restrict__ qkv, _Float16* __restrict__ vt) {
  int t = blockIdx.x * blockDim.x + threadIdx.x;
  if (t >= BB * 4 * DHH * LLEN) return;
  int k = t & (LLEN - 1), r = t >> 10, d = r & 31, bh = r >> 5;
  int b = bh >> 2, h = bh & 3;
  vt[t] = qkv[((size_t)b * LLEN + k) * 384 + 256 + h * DHH + d];
}

// ---------------------------------------------------------------------------
// Host-side orchestration
// ---------------------------------------------------------------------------
struct MambaP  { const float *W_in, *conv_w, *conv_b, *W_x, *W_dt, *b_dt, *A_log, *Dp, *W_out; };
struct ExpertP { const float *W1, *b1, *W2, *b2; };
struct LayerP  {
  MambaP mf, mb;
  const float *Wg, *bg; ExpertP ex[NEXP];
  const float *Wih, *Whh, *bih, *bhh;
  const float *Wqkv, *bqkv, *Wo, *bo;
  const float *n1g, *n1b, *n2g, *n2b, *n3g, *n3b;
};

static void launch_cvt(const float* src, _Float16* dst, int n, hipStream_t st) {
  cvt_f16_kernel<<<(n + 255) / 256, 256, 0, st>>>(src, dst, n);
}
static void launch_cvt_w(const float* src, _Float16* dst, int N, int K, int srcTrans,
                         hipStream_t st) {
  cvt_w_kernel<<<(N * K + 255) / 256, 256, 0, st>>>(src, dst, N, K, srcTrans);
}
static void launch_gemm(const _Float16* A, int lda, const _Float16* W, int ldw,
                        const float* bias, float* C, _Float16* Ch, int ldc,
                        int M, int N, int K,
                        int act, const float* gate, int gstride, int accum,
                        float alpha, hipStream_t st) {
  dim3 grid(M / 64, N / 32);
  gemm_wmma_kernel<<<grid, 128, 0, st>>>(A, lda, W, ldw, bias, C, Ch, ldc,
                                         M, N, K, act, gate, gstride, accum, alpha);
}

static void run_mamba(const MambaP& p, const float* xin, float* out,
                      float* bA, float* bU, float* bDl, float* bY, float* bI,
                      _Float16* hX, _Float16* hW, hipStream_t st) {
  launch_cvt(xin, hX, TOK * DD, st);
  launch_cvt_w(p.W_in, hW, 2 * DI_, DD, /*src (K,N)*/0, st);
  launch_gemm(hX, DD, hW, DD, nullptr, bA, nullptr, 2 * DI_, TOK, 2 * DI_, DD,
              0, nullptr, 0, 0, 1.0f, st);                                   // xr = x @ W_in
  conv_silu_kernel<<<(TOK * DI_) / 256, 256, 0, st>>>(bA, p.conv_w, p.conv_b, bU);
  xd_kernel<<<(TOK * 40 + 255) / 256, 256, 0, st>>>(bU, p.W_x, bI);
  delta_kernel<<<(TOK * DI_) / 256, 256, 0, st>>>(bI, p.W_dt, p.b_dt, bDl);
  scan_kernel<<<(BB * DI_) / 256, 256, 0, st>>>(bDl, bU, bI, p.A_log, bY);
  gate_y_kernel<<<(TOK * DI_) / 256, 256, 0, st>>>(bY, bU, bA, p.Dp, bY);
  launch_cvt(bY, hX, TOK * DI_, st);
  launch_cvt_w(p.W_out, hW, DD, DI_, /*src (K,N)*/0, st);
  launch_gemm(hX, DI_, hW, DI_, nullptr, out, nullptr, DD, TOK, DD, DI_,
              0, nullptr, 0, 0, 1.0f, st);                                   // out = y @ W_out
}

extern "C" void kernel_launch(void* const* d_in, const int* in_sizes, int n_in,
                              void* d_out, int out_size, void* d_ws, size_t ws_size,
                              hipStream_t stream) {
  // ---- input pointer walk (setup_inputs() insertion order) ----
  int idx = 0;
  auto nextp = [&]() { return (const float*)d_in[idx++]; };
  const float* x_in = nextp();
  LayerP layers[2];
  auto readMamba = [&](MambaP& m) {
    m.W_in = nextp(); m.conv_w = nextp(); m.conv_b = nextp(); m.W_x = nextp();
    m.W_dt = nextp(); m.b_dt = nextp(); m.A_log = nextp(); m.Dp = nextp(); m.W_out = nextp();
  };
  for (int l = 0; l < 2; ++l) {
    LayerP& L = layers[l];
    readMamba(L.mf); readMamba(L.mb);
    L.Wg = nextp(); L.bg = nextp();
    for (int e = 0; e < NEXP; ++e) {
      L.ex[e].W1 = nextp(); L.ex[e].b1 = nextp(); L.ex[e].W2 = nextp(); L.ex[e].b2 = nextp();
    }
    L.Wih = nextp(); L.Whh = nextp(); L.bih = nextp(); L.bhh = nextp();
    L.Wqkv = nextp(); L.bqkv = nextp(); L.Wo = nextp(); L.bo = nextp();
    L.n1g = nextp(); L.n1b = nextp();
    L.n2g = nextp(); L.n2b = nextp();
    L.n3g = nextp(); L.n3b = nextp();
  }

  // ---- workspace arena: f32 region then f16 region (offsets 32B aligned) ----
  float* wsf = (float*)d_ws;
  float* bA  = wsf;                          // (TOK,512) xr / gx (f32)
  float* bU  = bA  + (size_t)TOK * 512;      // (TOK,256) conv output u
  float* bDl = bU  + (size_t)TOK * 256;      // (TOK,256) delta
  float* bY  = bDl + (size_t)TOK * 256;      // (TOK,256) scan output y
  float* bE  = bY  + (size_t)TOK * 256;      // (TOK,128) dir-out / attn-out
  float* bF  = bE  + (size_t)TOK * 128;      // (TOK,128) residual branch accumulator
  float* bG  = bF  + (size_t)TOK * 128;      // (TOK,128) current x
  float* bH  = bG  + (size_t)TOK * 128;      // (TOK,128) reversed x / GRU hidden seq
  float* bI  = bH  + (size_t)TOK * 128;      // (TOK,64)  xd (stride 40) / gates (stride 4)
  _Float16* hX   = (_Float16*)(bI + (size_t)TOK * 64);  // (TOK,512) f16 activations
  _Float16* hH   = hX + (size_t)TOK * 512;              // (TOK,512) f16 MoE hidden
  _Float16* hW   = hH + (size_t)TOK * 512;              // up to 512*128 f16 weights (N,K)
  _Float16* hQKV = hW + (size_t)512 * 128;              // (TOK,384) f16 qkv
  _Float16* hP   = hQKV + (size_t)TOK * 384;            // (L,L) f16 attn probs (one bh)
  _Float16* hVT  = hP + (size_t)LLEN * LLEN;            // (32,32,1024) f16 V^T per bh
  float*    fS   = (float*)(hVT + (size_t)32 * DHH * LLEN); // (L,L) f32 scores (one bh)

  hipMemcpyAsync(bG, x_in, (size_t)TOK * DD * sizeof(float),
                 hipMemcpyDeviceToDevice, stream);

  const float att_scale = 0.176776695296637f;  // 1/sqrt(dh)

  for (int l = 0; l < 2; ++l) {
    LayerP& L = layers[l];

    // ---- bidirectional Mamba ----
    run_mamba(L.mf, bG, bF, bA, bU, bDl, bY, bI, hX, hW, stream);       // forward dir -> F
    reverse_copy_kernel<<<(TOK * DD) / 256, 256, 0, stream>>>(bG, bH);
    run_mamba(L.mb, bH, bE, bA, bU, bDl, bY, bI, hX, hW, stream);       // backward dir -> E
    add_reverse_kernel<<<(TOK * DD) / 256, 256, 0, stream>>>(bF, bE);   // F += rev(E)
    ln_kernel<<<TOK / 8, 256, 0, stream>>>(bG, bF, L.n1g, L.n1b, bG);   // x = LN(x + h)

    // ---- MoE: x converted once, reused by all experts; hidden kept in f16 ----
    moe_gate1_kernel<<<TOK / 256, 256, 0, stream>>>(bG, L.Wg, L.bg, bI);
    moe_gate2_kernel<<<(LLEN * NEXP + 255) / 256, 256, 0, stream>>>(bI);
    launch_cvt(bG, hX, TOK * DD, stream);
    for (int e = 0; e < NEXP; ++e) {
      launch_cvt_w(L.ex[e].W1, hW, 512, DD, 0, stream);
      launch_gemm(hX, DD, hW, DD, L.ex[e].b1, nullptr, hH, 512,
                  TOK, 512, DD, /*gelu*/1, nullptr, 0, 0, 1.0f, stream);
      launch_cvt_w(L.ex[e].W2, hW, DD, 512, 0, stream);
      launch_gemm(hH, 512, hW, 512, L.ex[e].b2, bF, nullptr, DD,
                  TOK, DD, 512, 0, /*gate*/bI + e, NEXP, e == 0 ? 0 : 1, 1.0f, stream);
    }
    ln_kernel<<<TOK / 8, 256, 0, stream>>>(bG, bF, L.n2g, L.n2b, bG);   // x = LN(x + moe)

    // ---- GRU ----
    launch_cvt(bG, hX, TOK * DD, stream);
    launch_cvt_w(L.Wih, hW, 384, DD, /*src (N,K)*/1, stream);
    launch_gemm(hX, DD, hW, DD, L.bih, bA, nullptr, 384,
                TOK, 384, DD, 0, nullptr, 0, 0, 1.0f, stream);          // gx
    gru_kernel<<<1, 1024, 0, stream>>>(bA, L.Whh, L.bhh, bH);

    // ---- MHA on GRU output (all three GEMMs on WMMA) ----
    launch_cvt(bH, hX, TOK * DD, stream);
    launch_cvt_w(L.Wqkv, hW, 384, DD, 1, stream);
    launch_gemm(hX, DD, hW, DD, L.bqkv, nullptr, hQKV, 384,
                TOK, 384, DD, 0, nullptr, 0, 0, 1.0f, stream);          // qkv (f16 out)
    vT_kernel<<<(BB * 4 * DHH * LLEN) / 256, 256, 0, stream>>>(hQKV, hVT);
    for (int bh = 0; bh < BB * 4; ++bh) {
      int b = bh >> 2, h = bh & 3;
      const _Float16* Q  = hQKV + ((size_t)b * LLEN) * 384 + h * DHH;
      const _Float16* Kp = hQKV + ((size_t)b * LLEN) * 384 + 128 + h * DHH;
      // S = alpha * Q K^T   (M=N=L, K=dh=32)
      launch_gemm(Q, 384, Kp, 384, nullptr, fS, nullptr, LLEN,
                  LLEN, LLEN, DHH, 0, nullptr, 0, 0, att_scale, stream);
      row_softmax_f16_kernel<<<LLEN / 8, 256, 0, stream>>>(fS, hP);
      // O = P V            (M=L, N=dh=32, K=L)
      launch_gemm(hP, LLEN, hVT + (size_t)bh * DHH * LLEN, LLEN, nullptr,
                  bE + ((size_t)b * LLEN) * DD + h * DHH, nullptr, DD,
                  LLEN, DHH, LLEN, 0, nullptr, 0, 0, 1.0f, stream);
    }
    launch_cvt(bE, hX, TOK * DD, stream);
    launch_cvt_w(L.Wo, hW, DD, DD, 1, stream);
    launch_gemm(hX, DD, hW, DD, L.bo, bF, nullptr, DD,
                TOK, DD, DD, 0, nullptr, 0, 0, 1.0f, stream);           // F = o@Wo^T + bo
    ln_kernel<<<TOK / 8, 256, 0, stream>>>(bG, bF, L.n3g, L.n3b, bG);   // x = LN(x + mha)
  }

  hipMemcpyAsync(d_out, bG, (size_t)TOK * DD * sizeof(float),
                 hipMemcpyDeviceToDevice, stream);
}
